// SpatioTemporalAttentionDiffusionNet_61186104099239
// MI455X (gfx1250) — matrix operand order repaired and checked by reading, not compiled, not run on hardware
//
#include <hip/hip_runtime.h>
#include <math.h>

#define B_    16
#define T_    24
#define N_    512
#define FE_   8
#define H_    64
#define FL_   12
#define PREJ_ 1024
#define KS_   3

#define BTN_  (B_*T_*N_)     /* 196608 */
#define BN_   (B_*N_)        /* 8192   */
#define S1_   (BTN_*H_)      /* 12582912 */
#define TH_   (T_*H_)        /* 1536   */

typedef _Float16 half16 __attribute__((ext_vector_type(16)));
typedef _Float16 half4v __attribute__((ext_vector_type(4)));
typedef float    float8 __attribute__((ext_vector_type(8)));

// ---------------------------------------------------------------------------
// WMMA core (wave32, v_wmma_f32_16x16x32_f16)
// ---------------------------------------------------------------------------
__device__ __forceinline__ float8 wmma16(half16 a, half16 b, float8 c) {
  return __builtin_amdgcn_wmma_f32_16x16x32_f16(false, a, false, b, (short)0, c,
                                                false, false);
}

// Tiles live in LDS in *fragment order*: a lane's 16 halves are contiguous,
// so one fragment = 2x ds_load_b128 and zero packing movs.
//
// A (16x32 per sub-tile, ISA 7.12.2):   lane<16 : slots 0-7 -> k, 8-15 -> k+8
//                                       lane>=16: slots 0-7 -> k+8, 8-15 -> k+16
// B (32x16 per col-group): same k order as A (lane<16: 0-7,16-23; lane>=16: 8-15,24-31)
__device__ __forceinline__ void a_slot(int base, int& row, int& k0) {
  int st  = base >> 9;           // row sub-tile (0: rows 0-15, 1: rows 16-31)
  int rem = base & 511;
  int ln  = rem >> 4;
  int s0  = rem & 15;            // multiple of 4
  row = st * 16 + (ln & 15);
  k0  = (ln < 16) ? ((s0 < 8) ? s0 : s0 + 8)
                  : ((s0 < 8) ? s0 + 8 : s0 + 16);
}
__device__ __forceinline__ void b_slot(int base, int& k0, int& col) {
  int g   = base >> 9;           // 16-wide column group
  int rem = base & 511;
  int ln  = rem >> 4;
  int s0  = rem & 15;
  col = g * 16 + (ln & 15);
  k0  = (ln < 16) ? ((s0 < 8) ? s0 : s0 + 8)
                  : ((s0 < 8) ? s0 + 8 : s0 + 16);
}
__device__ __forceinline__ half16 read_frag(const _Float16* p, int lane) {
  return *(const half16*)(p + lane * 16);
}

// branch-free activations (single v_exp_f32 each)
__device__ __forceinline__ float fsigm(float x) { return 1.f / (1.f + __expf(-x)); }
__device__ __forceinline__ float ftanh(float x) { return 1.f - 2.f / (__expf(2.f * x) + 1.f); }

// ---------------------------------------------------------------------------
// 1. projection: X_aug[.,9] @ W_proj[9,64] + b   (K=9 -> VALU)
// ---------------------------------------------------------------------------
__global__ __launch_bounds__(256) void k_proj(const float* __restrict__ X,
                                              const unsigned char* __restrict__ mask,
                                              const float* __restrict__ Wp,
                                              const float* __restrict__ bp,
                                              float* __restrict__ out) {
  int idx = blockIdx.x * 256 + threadIdx.x;     // over S1_
  int row = idx >> 6, o = idx & 63;
  const float* xr = X + row * FE_;
  float acc = bp[o];
#pragma unroll
  for (int f = 0; f < FE_; ++f) acc += xr[f] * Wp[f * H_ + o];
  float mc = mask[row * FE_] ? 1.f : 0.f;
  acc += mc * Wp[FE_ * H_ + o];
  out[idx] = acc;
}

// ---------------------------------------------------------------------------
// 2. gated temporal conv (im2col K = KS*H = 192), 32x64 tile, dual B
//    out_mode 0: [B,T,N,H]   out_mode 1: [B,N,T,H]
// ---------------------------------------------------------------------------
__global__ __launch_bounds__(128) void k_gated_tconv(
    const float* __restrict__ in, const float* __restrict__ Wc,
    const float* __restrict__ bc, const float* __restrict__ Wg,
    const float* __restrict__ bg, float* __restrict__ out, int out_mode) {
  __shared__ __align__(32) _Float16 As[1024];
  __shared__ __align__(32) _Float16 Bc[2048];
  __shared__ __align__(32) _Float16 Bg[2048];
  int tid = threadIdx.x, lane = tid & 31, wave = tid >> 5;
  int r0  = blockIdx.x * 32;                    // 32 rows share (b,t)
  int b   = r0 / (T_ * N_);
  int rem = r0 - b * (T_ * N_);
  int t   = rem / N_;
  int n0  = rem - t * N_;
  float8 aC0 = {0,0,0,0,0,0,0,0}, aC1 = {0,0,0,0,0,0,0,0};
  float8 aG0 = {0,0,0,0,0,0,0,0}, aG1 = {0,0,0,0,0,0,0,0};
  for (int kc = 0; kc < 6; ++kc) {
    int tap = kc >> 1, i0 = (kc & 1) * 32;
    int tsrc = t + tap - (KS_ - 1);
    const float* src = in + ((b * T_ + tsrc) * N_ + n0) * H_ + i0;
    if (tsrc >= 0) {                            // block-uniform branch
#pragma unroll
      for (int u = 0; u < 2; ++u) {             // A: 1024 halves
        int base = (tid + u * 128) * 4;
        int row, k0; a_slot(base, row, k0);
        half4v h;
#pragma unroll
        for (int ii = 0; ii < 4; ++ii)
          h[ii] = (_Float16)src[row * H_ + k0 + ii];
        *(half4v*)(As + base) = h;
      }
    } else {
      half4v z = {(_Float16)0.f, (_Float16)0.f, (_Float16)0.f, (_Float16)0.f};
#pragma unroll
      for (int u = 0; u < 2; ++u) *(half4v*)(As + (tid + u * 128) * 4) = z;
    }
#pragma unroll
    for (int u = 0; u < 4; ++u) {               // Bc/Bg: 2048 halves each
      int base = (tid + u * 128) * 4;
      int k0, col; b_slot(base, k0, col);
      half4v hc, hg;
#pragma unroll
      for (int ii = 0; ii < 4; ++ii) {
        int wi = ((col << 6) + i0 + k0 + ii) * KS_ + tap;   // W[o][i][k]
        hc[ii] = (_Float16)Wc[wi];
        hg[ii] = (_Float16)Wg[wi];
      }
      *(half4v*)(Bc + base) = hc;
      *(half4v*)(Bg + base) = hg;
    }
    __syncthreads();
    half16 a0 = read_frag(As, lane);
    half16 a1 = read_frag(As + 512, lane);
    half16 bcf = read_frag(Bc + wave * 512, lane);
    half16 bgf = read_frag(Bg + wave * 512, lane);
    aC0 = wmma16(a0, bcf, aC0);
    aC1 = wmma16(a1, bcf, aC1);
    aG0 = wmma16(a0, bgf, aG0);
    aG1 = wmma16(a1, bgf, aG1);
    __syncthreads();
  }
  int col = wave * 16 + (lane & 15);
  float bC = bc[col], bG = bg[col];
  int mb = (lane < 16) ? 0 : 8;
#pragma unroll
  for (int v = 0; v < 8; ++v) {
    int m = v + mb;
#pragma unroll
    for (int st = 0; st < 2; ++st) {
      float cc = (st == 0) ? aC0[v] : aC1[v];
      float gg = (st == 0) ? aG0[v] : aG1[v];
      float val = ftanh(cc + bC) * fsigm(gg + bG);
      int n = n0 + st * 16 + m;
      int oidx = (out_mode == 0) ? ((b * T_ + t) * N_ + n) * H_ + col
                                 : ((b * N_ + n) * T_ + t) * H_ + col;
      out[oidx] = val;
    }
  }
}

// ---------------------------------------------------------------------------
// 3. Q/K projections ([B,N] rows, K=T*H gather), 32x64 tile, dual B
// ---------------------------------------------------------------------------
__global__ __launch_bounds__(128) void k_qk(const float* __restrict__ xg,
                                            const float* __restrict__ Wq,
                                            const float* __restrict__ bq,
                                            const float* __restrict__ Wk,
                                            const float* __restrict__ bk,
                                            float* __restrict__ Q,
                                            float* __restrict__ K) {
  __shared__ __align__(32) _Float16 As[1024];
  __shared__ __align__(32) _Float16 BQ[2048];
  __shared__ __align__(32) _Float16 BK[2048];
  int tid = threadIdx.x, lane = tid & 31, wave = tid >> 5;
  int r0 = blockIdx.x * 32;
  int b = r0 >> 9, n0 = r0 & 511;
  float8 q0 = {0,0,0,0,0,0,0,0}, q1 = {0,0,0,0,0,0,0,0};
  float8 k0a = {0,0,0,0,0,0,0,0}, k1a = {0,0,0,0,0,0,0,0};
  for (int kc = 0; kc < TH_ / 32; ++kc) {
    int c0 = kc * 32, t = kc >> 1, h0 = (kc & 1) * 32;
    const float* src = xg + ((b * T_ + t) * N_ + n0) * H_ + h0;
#pragma unroll
    for (int u = 0; u < 2; ++u) {
      int base = (tid + u * 128) * 4;
      int row, kk; a_slot(base, row, kk);
      half4v h;
#pragma unroll
      for (int ii = 0; ii < 4; ++ii) h[ii] = (_Float16)src[row * H_ + kk + ii];
      *(half4v*)(As + base) = h;
    }
#pragma unroll
    for (int u = 0; u < 4; ++u) {
      int base = (tid + u * 128) * 4;
      int kk, col; b_slot(base, kk, col);
      half4v hq, hk;
#pragma unroll
      for (int ii = 0; ii < 4; ++ii) {
        int wi = (c0 + kk + ii) * H_ + col;
        hq[ii] = (_Float16)Wq[wi];
        hk[ii] = (_Float16)Wk[wi];
      }
      *(half4v*)(BQ + base) = hq;
      *(half4v*)(BK + base) = hk;
    }
    __syncthreads();
    half16 a0 = read_frag(As, lane);
    half16 a1 = read_frag(As + 512, lane);
    half16 bq_f = read_frag(BQ + wave * 512, lane);
    half16 bk_f = read_frag(BK + wave * 512, lane);
    q0 = wmma16(a0, bq_f, q0);
    q1 = wmma16(a1, bq_f, q1);
    k0a = wmma16(a0, bk_f, k0a);
    k1a = wmma16(a1, bk_f, k1a);
    __syncthreads();
  }
  int col = wave * 16 + (lane & 15);
  float bqv = bq[col], bkv = bk[col];
  int mb = (lane < 16) ? 0 : 8;
#pragma unroll
  for (int v = 0; v < 8; ++v) {
    int m = v + mb;
    Q[(r0 + m) * H_ + col] = q0[v] + bqv;
    K[(r0 + m) * H_ + col] = k0a[v] + bkv;
    Q[(r0 + 16 + m) * H_ + col] = q1[v] + bqv;
    K[(r0 + 16 + m) * H_ + col] = k1a[v] + bkv;
  }
}

// ---------------------------------------------------------------------------
// 4. scores = Q @ K^T / sqrt(H), 32x64 tile
// ---------------------------------------------------------------------------
__global__ __launch_bounds__(128) void k_scores(const float* __restrict__ Q,
                                                const float* __restrict__ K,
                                                float* __restrict__ S) {
  __shared__ __align__(32) _Float16 As[1024];
  __shared__ __align__(32) _Float16 Bs[2048];
  int tid = threadIdx.x, lane = tid & 31, wave = tid >> 5;
  int r0 = blockIdx.x * 32;
  int b = r0 >> 9;
  int m0 = blockIdx.y * 64;
  float8 s0 = {0,0,0,0,0,0,0,0}, s1 = {0,0,0,0,0,0,0,0};
  for (int kc = 0; kc < 2; ++kc) {
    int c0 = kc * 32;
#pragma unroll
    for (int u = 0; u < 2; ++u) {
      int base = (tid + u * 128) * 4;
      int row, kk; a_slot(base, row, kk);
      half4v h;
#pragma unroll
      for (int ii = 0; ii < 4; ++ii)
        h[ii] = (_Float16)Q[(r0 + row) * H_ + c0 + kk + ii];
      *(half4v*)(As + base) = h;
    }
#pragma unroll
    for (int u = 0; u < 4; ++u) {
      int base = (tid + u * 128) * 4;
      int kk, col; b_slot(base, kk, col);
      const float* kr = K + ((b << 9) + m0 + col) * H_ + c0 + kk;  // K^T gather
      half4v h;
#pragma unroll
      for (int ii = 0; ii < 4; ++ii) h[ii] = (_Float16)kr[ii];
      *(half4v*)(Bs + base) = h;
    }
    __syncthreads();
    half16 bf = read_frag(Bs + wave * 512, lane);
    s0 = wmma16(read_frag(As, lane), bf, s0);
    s1 = wmma16(read_frag(As + 512, lane), bf, s1);
    __syncthreads();
  }
  int col = m0 + wave * 16 + (lane & 15);
  int mb = (lane < 16) ? 0 : 8;
#pragma unroll
  for (int v = 0; v < 8; ++v) {
    int m = v + mb;
    S[(r0 + m) * N_ + col] = s0[v] * 0.125f;
    S[(r0 + 16 + m) * N_ + col] = s1[v] * 0.125f;
  }
}

// ---------------------------------------------------------------------------
// 5. graph-masked softmax (512-wide rows, in-place -> P)
// ---------------------------------------------------------------------------
__global__ __launch_bounds__(256) void k_softmax(float* __restrict__ S,
                                                 const float* __restrict__ adj) {
  __shared__ float red[256];
  int row = blockIdx.x;           // b*N + n
  int n = row & 511;
  int tid = threadIdx.x;
  int base = row * N_;
  int m1 = tid, m2 = tid + 256;
  bool v1 = (adj[n * N_ + m1] > 0.f) || (m1 == n);
  bool v2 = (adj[n * N_ + m2] > 0.f) || (m2 == n);
  float s1 = v1 ? S[base + m1] : -3.0e38f;
  float s2 = v2 ? S[base + m2] : -3.0e38f;
  red[tid] = fmaxf(s1, s2);
  __syncthreads();
  for (int s = 128; s > 0; s >>= 1) {
    if (tid < s) red[tid] = fmaxf(red[tid], red[tid + s]);
    __syncthreads();
  }
  float mx = red[0];
  __syncthreads();
  float e1 = v1 ? __expf(s1 - mx) : 0.f;
  float e2 = v2 ? __expf(s2 - mx) : 0.f;
  red[tid] = e1 + e2;
  __syncthreads();
  for (int s = 128; s > 0; s >>= 1) {
    if (tid < s) red[tid] += red[tid + s];
    __syncthreads();
  }
  float inv = 1.f / red[0];
  S[base + m1] = e1 * inv;
  S[base + m2] = e2 * inv;
}

// 6. fsum = bsum = x
__global__ __launch_bounds__(256) void k_copy2(const float* __restrict__ src,
                                               float* __restrict__ a,
                                               float* __restrict__ b) {
  int i = blockIdx.x * 256 + threadIdx.x;
  float v = src[i];
  a[i] = v;
  b[i] = v;
}

// ---------------------------------------------------------------------------
// 7. diffusion hop: vout[b,t,n,:] = P[b] @ vin[b,t,:,:], fused alpha-axpy
// ---------------------------------------------------------------------------
__global__ __launch_bounds__(128) void k_hop(const float* __restrict__ P,
                                             const float* __restrict__ vin,
                                             float* __restrict__ vout,
                                             float* __restrict__ fsum,
                                             float* __restrict__ bsum,
                                             const float* __restrict__ af_p,
                                             const float* __restrict__ ab_p,
                                             int hop) {
  __shared__ __align__(32) _Float16 As[1024];
  __shared__ __align__(32) _Float16 Bs[2048];
  int tid = threadIdx.x, lane = tid & 31, wave = tid >> 5;
  int r0 = blockIdx.x * 32;
  int b = r0 / (T_ * N_);
  int rem = r0 - b * (T_ * N_);
  int t = rem / N_;
  int n0 = rem - t * N_;
  const float* Pb = P + b * N_ * N_;
  const float* vb = vin + (b * T_ + t) * N_ * H_;
  float8 acc0 = {0,0,0,0,0,0,0,0}, acc1 = {0,0,0,0,0,0,0,0};
  for (int kc = 0; kc < 16; ++kc) {
    int c0 = kc * 32;
#pragma unroll
    for (int u = 0; u < 2; ++u) {
      int base = (tid + u * 128) * 4;
      int row, kk; a_slot(base, row, kk);
      half4v h;
#pragma unroll
      for (int ii = 0; ii < 4; ++ii)
        h[ii] = (_Float16)Pb[(n0 + row) * N_ + c0 + kk + ii];
      *(half4v*)(As + base) = h;
    }
#pragma unroll
    for (int u = 0; u < 4; ++u) {
      int base = (tid + u * 128) * 4;
      int kk, col; b_slot(base, kk, col);
      half4v h;
#pragma unroll
      for (int ii = 0; ii < 4; ++ii)
        h[ii] = (_Float16)vb[(c0 + kk + ii) * H_ + col];
      *(half4v*)(Bs + base) = h;
    }
    __syncthreads();
    half16 bf = read_frag(Bs + wave * 512, lane);
    acc0 = wmma16(read_frag(As, lane), bf, acc0);
    acc1 = wmma16(read_frag(As + 512, lane), bf, acc1);
    __syncthreads();
  }
  float af = af_p[hop], ab = ab_p[hop];
  int col = wave * 16 + (lane & 15);
  int mb = (lane < 16) ? 0 : 8;
#pragma unroll
  for (int v = 0; v < 8; ++v) {
    int m = v + mb;
#pragma unroll
    for (int st = 0; st < 2; ++st) {
      float o = (st == 0) ? acc0[v] : acc1[v];
      int oidx = (r0 + st * 16 + m) * H_ + col;
      vout[oidx] = o;
      fsum[oidx] += af * o;
      bsum[oidx] += ab * o;
    }
  }
}

// ---------------------------------------------------------------------------
// 8. generic GEMM: C[M,N] (+)= act(A[M,K] @ B[K,N] + bias)
//    REQUIRES: M % 32 == 0, K % 32 == 0.  N arbitrary (B cols clamped; only
//    in-range output columns are stored, so no zero-fill needed).
//    act: 0 none, 1 relu, 2 tanh.  accflag: add existing C before act.
// ---------------------------------------------------------------------------
__global__ __launch_bounds__(128) void k_gemm(const float* __restrict__ A,
                                              const float* __restrict__ Bw,
                                              const float* __restrict__ bias,
                                              float* __restrict__ C, int M,
                                              int K, int N, int act,
                                              int accflag) {
  __shared__ __align__(32) _Float16 As[1024];
  __shared__ __align__(32) _Float16 Bs[2048];
  int tid = threadIdx.x, lane = tid & 31, wave = tid >> 5;
  int r0 = blockIdx.x * 32;
  int n0 = blockIdx.y * 64;
  float8 acc0 = {0,0,0,0,0,0,0,0}, acc1 = {0,0,0,0,0,0,0,0};
  int nk = K >> 5;
  for (int kc = 0; kc < nk; ++kc) {
    int c0 = kc << 5;
#pragma unroll
    for (int u = 0; u < 2; ++u) {
      int base = (tid + u * 128) * 4;
      int row, kk; a_slot(base, row, kk);
      const float* ar = A + (r0 + row) * K + c0 + kk;
      half4v h;
#pragma unroll
      for (int ii = 0; ii < 4; ++ii) h[ii] = (_Float16)ar[ii];
      *(half4v*)(As + base) = h;
    }
#pragma unroll
    for (int u = 0; u < 4; ++u) {
      int base = (tid + u * 128) * 4;
      int kk, col; b_slot(base, kk, col);
      int n = n0 + col;
      int nc = (n < N) ? n : (N - 1);           // branch-free clamp
      half4v h;
#pragma unroll
      for (int ii = 0; ii < 4; ++ii)
        h[ii] = (_Float16)Bw[(c0 + kk + ii) * N + nc];
      *(half4v*)(Bs + base) = h;
    }
    __syncthreads();
    half16 bf = read_frag(Bs + wave * 512, lane);
    acc0 = wmma16(read_frag(As, lane), bf, acc0);
    acc1 = wmma16(read_frag(As + 512, lane), bf, acc1);
    __syncthreads();
  }
  int col = n0 + wave * 16 + (lane & 15);
  if (col < N) {
    float bb = bias ? bias[col] : 0.f;
    int mb = (lane < 16) ? 0 : 8;
#pragma unroll
    for (int v = 0; v < 8; ++v) {
      int m = v + mb;
#pragma unroll
      for (int st = 0; st < 2; ++st) {
        int oidx = (r0 + st * 16 + m) * N + col;
        float val = ((st == 0) ? acc0[v] : acc1[v]) + bb +
                    (accflag ? C[oidx] : 0.f);
        if (act == 1) val = fmaxf(val, 0.f);
        else if (act == 2) val = ftanh(val);
        C[oidx] = val;
      }
    }
  }
}

// ---------------------------------------------------------------------------
// 9. imputed_hidden = imputed @ Wi + bi  (K=8 -> VALU, rows (b,n,t))
// ---------------------------------------------------------------------------
__global__ __launch_bounds__(256) void k_wi(const float* __restrict__ imp,
                                            const float* __restrict__ Wi,
                                            const float* __restrict__ bi,
                                            float* __restrict__ out) {
  int idx = blockIdx.x * 256 + threadIdx.x;     // over S1_
  int row = idx >> 6, o = idx & 63;
  const float* xr = imp + row * FE_;
  float acc = bi[o];
#pragma unroll
  for (int f = 0; f < FE_; ++f) acc += xr[f] * Wi[f * H_ + o];
  out[idx] = acc;
}

// ---------------------------------------------------------------------------
// 10. mask_red[b,n] = any_{t,fe} mask
// ---------------------------------------------------------------------------
__global__ __launch_bounds__(256) void k_maskred(const unsigned char* __restrict__ mask,
                                                 float* __restrict__ R) {
  int i = blockIdx.x * 256 + threadIdx.x;   // BN_
  int b = i >> 9, n = i & 511;
  unsigned long long any = 0ull;
  for (int t = 0; t < T_; ++t) {
    const unsigned long long* p =
        (const unsigned long long*)(mask + (size_t)((b * T_ + t) * N_ + n) * FE_);
    any |= *p;
  }
  R[i] = any ? 1.f : 0.f;
}

// 11. mixed = mr*FwdOut + (1-mr)*imputed_hidden
__global__ __launch_bounds__(256) void k_mixed(const float* __restrict__ R,
                                               const float* __restrict__ F,
                                               const float* __restrict__ I,
                                               float* __restrict__ O) {
  int idx = blockIdx.x * 256 + threadIdx.x;   // S1_
  float mr = R[idx / TH_];
  O[idx] = mr * F[idx] + (1.f - mr) * I[idx];
}

// 12. output transposes
__global__ __launch_bounds__(256) void k_out_imputed(const float* __restrict__ imp,
                                                     float* __restrict__ out) {
  int idx = blockIdx.x * 256 + threadIdx.x;  // [B,T,N,FE]
  int fe = idx & 7, r = idx >> 3;
  int n = r & 511, r2 = r >> 9;
  int t = r2 % T_, b = r2 / T_;
  out[idx] = imp[((b * N_ + n) * T_ + t) * FE_ + fe];
}
__global__ __launch_bounds__(256) void k_out_forecast(const float* __restrict__ fc,
                                                      float* __restrict__ out) {
  int idx = blockIdx.x * 256 + threadIdx.x;  // [B,FL,N,FE]
  int fe = idx & 7, r = idx >> 3;
  int n = r & 511, r2 = r >> 9;
  int f = r2 % FL_, b = r2 / FL_;
  out[idx] = fc[((b * N_ + n) * FL_ + f) * FE_ + fe];
}

// ---------------------------------------------------------------------------
extern "C" void kernel_launch(void* const* d_in, const int* in_sizes, int n_in,
                              void* d_out, int out_size, void* d_ws,
                              size_t ws_size, hipStream_t stream) {
  (void)in_sizes; (void)n_in; (void)out_size; (void)ws_size;
  const float* X      = (const float*)d_in[0];
  const float* adj    = (const float*)d_in[1];
  const unsigned char* mask = (const unsigned char*)d_in[2];
  const float* W_proj = (const float*)d_in[3];
  const float* b_proj = (const float*)d_in[4];
  const float* t1_Wc  = (const float*)d_in[5];
  const float* t1_bc  = (const float*)d_in[6];
  const float* t1_Wg  = (const float*)d_in[7];
  const float* t1_bg  = (const float*)d_in[8];
  const float* tf_Wc  = (const float*)d_in[9];
  const float* tf_bc  = (const float*)d_in[10];
  const float* tf_Wg  = (const float*)d_in[11];
  const float* tf_bg  = (const float*)d_in[12];
  const float* tb_Wc  = (const float*)d_in[13];
  const float* tb_bc  = (const float*)d_in[14];
  const float* tb_Wg  = (const float*)d_in[15];
  const float* tb_bg  = (const float*)d_in[16];
  const float* Wq     = (const float*)d_in[17];
  const float* bq     = (const float*)d_in[18];
  const float* Wk     = (const float*)d_in[19];
  const float* bk     = (const float*)d_in[20];
  const float* alpha_f = (const float*)d_in[21];
  const float* alpha_b = (const float*)d_in[22];
  const float* Wf     = (const float*)d_in[23];
  const float* bfv    = (const float*)d_in[24];
  const float* Wb     = (const float*)d_in[25];
  const float* bbv    = (const float*)d_in[26];
  const float* Wi     = (const float*)d_in[27];
  const float* bi     = (const float*)d_in[28];
  const float* dW1    = (const float*)d_in[29];
  const float* db1    = (const float*)d_in[30];
  const float* dW2    = (const float*)d_in[31];
  const float* db2    = (const float*)d_in[32];
  const float* fW1    = (const float*)d_in[33];
  const float* fb1    = (const float*)d_in[34];
  const float* fW2    = (const float*)d_in[35];
  const float* fb2    = (const float*)d_in[36];

  float* ws = (float*)d_ws;
  float* W0 = ws;                         // proj out -> Fwd -> imputed_hidden
  float* W1 = ws + (size_t)1 * S1_;       // x (tconv1) -> BwdOut -> forecast
  float* W2 = ws + (size_t)2 * S1_;       // fsum -> h -> h2
  float* W3 = ws + (size_t)3 * S1_;       // bsum -> imputed
  float* W4 = ws + (size_t)4 * S1_;       // hop ping -> FwdOut
  float* W5 = ws + (size_t)5 * S1_;       // hop pong -> Bwd -> mixed
  float* Qb = ws + (size_t)6 * S1_;
  float* Kb = Qb + (size_t)BN_ * H_;
  float* Pb = Kb + (size_t)BN_ * H_;
  float* Rb = Pb + (size_t)B_ * N_ * N_;
  float* out_imp = (float*)d_out;
  float* out_fc  = out_imp + (size_t)B_ * T_ * N_ * FE_;

  // projection (K=9, VALU)
  k_proj<<<S1_ / 256, 256, 0, stream>>>(X, mask, W_proj, b_proj, W0);
  // tconv1 -> x
  k_gated_tconv<<<BTN_ / 32, 128, 0, stream>>>(W0, t1_Wc, t1_bc, t1_Wg, t1_bg, W1, 0);
  // attention
  k_qk<<<BN_ / 32, 128, 0, stream>>>(W1, Wq, bq, Wk, bk, Qb, Kb);
  k_scores<<<dim3(BN_ / 32, N_ / 64), 128, 0, stream>>>(Qb, Kb, Pb);
  k_softmax<<<BN_, 256, 0, stream>>>(Pb, adj);
  // diffusion (v_f == v_b: single chain, dual alpha accumulation)
  k_copy2<<<S1_ / 256, 256, 0, stream>>>(W1, W2, W3);
  k_hop<<<BTN_ / 32, 128, 0, stream>>>(Pb, W1, W4, W2, W3, alpha_f, alpha_b, 0);
  k_hop<<<BTN_ / 32, 128, 0, stream>>>(Pb, W4, W5, W2, W3, alpha_f, alpha_b, 1);
  k_hop<<<BTN_ / 32, 128, 0, stream>>>(Pb, W5, W4, W2, W3, alpha_f, alpha_b, 2);
  // Fwd / Bwd (tanh)
  k_gemm<<<dim3(BTN_ / 32, 1), 128, 0, stream>>>(W2, Wf, bfv, W0, BTN_, H_, H_, 2, 0);
  k_gemm<<<dim3(BTN_ / 32, 1), 128, 0, stream>>>(W3, Wb, bbv, W5, BTN_, H_, H_, 2, 0);
  // output tconvs -> [B,N,T,H]
  k_gated_tconv<<<BTN_ / 32, 128, 0, stream>>>(W0, tf_Wc, tf_bc, tf_Wg, tf_bg, W4, 1);
  k_gated_tconv<<<BTN_ / 32, 128, 0, stream>>>(W5, tb_Wc, tb_bc, tb_Wg, tb_bg, W1, 1);
  // h = relu(concat(Fwd_out,Bwd_out) @ dW1 + db1) via two K-passes
  k_gemm<<<dim3(BN_ / 32, PREJ_ / 64), 128, 0, stream>>>(W4, dW1, nullptr, W2,
                                                         BN_, TH_, PREJ_, 0, 0);
  k_gemm<<<dim3(BN_ / 32, PREJ_ / 64), 128, 0, stream>>>(W1, dW1 + TH_ * PREJ_,
                                                         db1, W2, BN_, TH_, PREJ_, 1, 1);
  // imputed = relu(h @ dW2 + db2)  [B,N,T*FE]
  k_gemm<<<dim3(BN_ / 32, (T_ * FE_) / 64), 128, 0, stream>>>(W2, dW2, db2, W3,
                                                              BN_, PREJ_, T_ * FE_, 1, 0);
  // imputed_hidden = imputed @ Wi + bi  (K=8 -> VALU) -> [B,N,T,H]
  k_wi<<<S1_ / 256, 256, 0, stream>>>(W3, Wi, bi, W0);
  // gate mix
  k_maskred<<<BN_ / 256, 256, 0, stream>>>(mask, Rb);
  k_mixed<<<S1_ / 256, 256, 0, stream>>>(Rb, W4, W0, W5);
  // forecast head
  k_gemm<<<dim3(BN_ / 32, PREJ_ / 64), 128, 0, stream>>>(W5, fW1, fb1, W2, BN_,
                                                         TH_, PREJ_, 1, 0);
  k_gemm<<<dim3(BN_ / 32, 2), 128, 0, stream>>>(W2, fW2, fb2, W1, BN_, PREJ_,
                                                FL_ * FE_, 1, 0);
  // outputs
  k_out_imputed<<<(B_ * T_ * N_ * FE_) / 256, 256, 0, stream>>>(W3, out_imp);
  k_out_forecast<<<(B_ * FL_ * N_ * FE_) / 256, 256, 0, stream>>>(W1, out_fc);
}
